// Attention_57741540327517
// MI455X (gfx1250) — matrix operand order repaired and checked by reading, not compiled, not run on hardware
//
#include <hip/hip_runtime.h>
#include <hip/hip_bf16.h>

typedef __attribute__((ext_vector_type(16))) __bf16 v16bf;
typedef __attribute__((ext_vector_type(8)))  float  v8f;

// ---------- scalar helpers ----------

__device__ __forceinline__ unsigned short f2bf(float f) {
  unsigned int u = __float_as_uint(f);
  unsigned int r = u + 0x7FFFu + ((u >> 16) & 1u);   // round-to-nearest-even
  return (unsigned short)(r >> 16);
}
__device__ __forceinline__ unsigned int pack2bf(float lo, float hi) {
  return (unsigned int)f2bf(lo) | ((unsigned int)f2bf(hi) << 16);
}

__device__ __forceinline__ v8f wmma_bf16(v16bf a, v16bf b, v8f c) {
  return __builtin_amdgcn_wmma_f32_16x16x32_bf16(false, a, false, b, (short)0, c,
                                                 false, false);
}

// ---------- CDNA5 async memory->LDS copy (ASYNCcnt-tracked) ----------

__device__ __forceinline__ void async_copy16(unsigned int lds_off, const void* gptr) {
  asm volatile("global_load_async_to_lds_b128 %0, %1, off"
               :: "v"(lds_off), "v"((unsigned long long)(uintptr_t)gptr)
               : "memory");
}
__device__ __forceinline__ void wait_async_0() {
  asm volatile("s_wait_asynccnt 0x0" ::: "memory");
}
__device__ __forceinline__ void wait_async_4() {
  asm volatile("s_wait_asynccnt 0x4" ::: "memory");
}
__device__ __forceinline__ unsigned int lds_off_of(const void* p) {
  return (unsigned int)(uintptr_t)p;   // addrspace(3) offset lives in low 32 bits
}

// ---------- WMMA fragment loaders (contiguous k = 16h + s, same order in A and B) ----------

// A fragment (16x32, MxK) from LDS tile stored [m][k], row stride ldk (bf16 units).
__device__ __forceinline__ v16bf load_a_frag(const unsigned short* base, int ldk,
                                             int m0, int k0) {
  const int lane = threadIdx.x & 31;
  const int m  = m0 + (lane & 15);
  const int kb = (lane & 16) ? 16 : 0;
  const unsigned short* row = base + m * ldk + k0;
  union { v16bf v; unsigned int d[8]; } a;
#pragma unroll
  for (int i = 0; i < 8; ++i) {
    const int kk = kb + 2 * i;
    a.d[i] = *(const unsigned int*)(row + kk);
  }
  return a.v;
}

// B fragment (32x16, KxN) from LDS tile stored [n][k] (K-contiguous rows).
__device__ __forceinline__ v16bf load_b_frag_nk(const unsigned short* base, int ldk,
                                                int n0, int k0) {
  const int lane = threadIdx.x & 31;
  const int n  = n0 + (lane & 15);
  const int kb = (lane & 16) ? 16 : 0;
  const unsigned short* row = base + n * ldk + k0 + kb;
  union { v16bf v; unsigned int d[8]; } b;
#pragma unroll
  for (int i = 0; i < 8; ++i) b.d[i] = *(const unsigned int*)(row + 2 * i);
  return b.v;
}

// ---------- kernel 0: one-time fp32 -> bf16 conversion ----------

__global__ __launch_bounds__(256) void f32_to_bf16(const float* __restrict__ in,
                                                   unsigned short* __restrict__ out,
                                                   int n8) {
  const int i = blockIdx.x * 256 + threadIdx.x;
  if (i < n8) {
    const float4* p = (const float4*)(in + (size_t)i * 8);
    float4 a = p[0], b = p[1];
    unsigned int* o = (unsigned int*)(out + (size_t)i * 8);
    o[0] = pack2bf(a.x, a.y); o[1] = pack2bf(a.z, a.w);
    o[2] = pack2bf(b.x, b.y); o[3] = pack2bf(b.z, b.w);
  }
}

// ---------- GEMM: C[M,N] = A[M,K] @ W[N,K]^T + bias, all-bf16 inputs ----------
// 64x64x64 block stage, 8 waves (4Mx2N), double-buffered async LDS staging.

template <bool BF16OUT>
__global__ __launch_bounds__(256) void gemm_bf16_async(
    const unsigned short* __restrict__ A, const unsigned short* __restrict__ W,
    const float* __restrict__ bias, void* __restrict__ Cout,
    int M, int N, int K) {
  __shared__ unsigned short As[2][64 * 64];
  __shared__ unsigned short Bs[2][64 * 64];

  const int tid = threadIdx.x;
  const int m0 = blockIdx.y * 64;
  const int n0 = blockIdx.x * 64;
  const int w  = tid >> 5;
  const int wm = (w >> 1) * 16;
  const int wn = (w & 1) * 32;
  const int row = tid >> 2;           // 0..63
  const int s0  = (tid & 3) * 16;     // 2x 16B chunks per 128B row
  const int s1  = s0 + 8;

  const unsigned int a_lds[2][2] = {
      {lds_off_of(&As[0][row * 64 + s0]), lds_off_of(&As[0][row * 64 + s1])},
      {lds_off_of(&As[1][row * 64 + s0]), lds_off_of(&As[1][row * 64 + s1])}};
  const unsigned int b_lds[2][2] = {
      {lds_off_of(&Bs[0][row * 64 + s0]), lds_off_of(&Bs[0][row * 64 + s1])},
      {lds_off_of(&Bs[1][row * 64 + s0]), lds_off_of(&Bs[1][row * 64 + s1])}};
  const unsigned short* agp = A + (size_t)(m0 + row) * K;
  const unsigned short* bgp = W + (size_t)(n0 + row) * K;

  v8f acc0 = {}; v8f acc1 = {};

  // prime buffer 0 (4 async ops per wave)
  async_copy16(a_lds[0][0], agp + s0);
  async_copy16(a_lds[0][1], agp + s1);
  async_copy16(b_lds[0][0], bgp + s0);
  async_copy16(b_lds[0][1], bgp + s1);

  const int NT = K >> 6;
  for (int t = 0; t < NT; ++t) {
    const int cur = t & 1;
    if (t + 1 < NT) {
      const int k1 = (t + 1) * 64;
      async_copy16(a_lds[cur ^ 1][0], agp + k1 + s0);
      async_copy16(a_lds[cur ^ 1][1], agp + k1 + s1);
      async_copy16(b_lds[cur ^ 1][0], bgp + k1 + s0);
      async_copy16(b_lds[cur ^ 1][1], bgp + k1 + s1);
      wait_async_4();                 // drain old stage, keep new 4 in flight
    } else {
      wait_async_0();
    }
    __syncthreads();

    v16bf a0  = load_a_frag(As[cur], 64, wm, 0);
    v16bf a1  = load_a_frag(As[cur], 64, wm, 32);
    v16bf b00 = load_b_frag_nk(Bs[cur], 64, wn, 0);
    v16bf b01 = load_b_frag_nk(Bs[cur], 64, wn, 32);
    v16bf b10 = load_b_frag_nk(Bs[cur], 64, wn + 16, 0);
    v16bf b11 = load_b_frag_nk(Bs[cur], 64, wn + 16, 32);
    acc0 = wmma_bf16(a0, b00, acc0);
    acc0 = wmma_bf16(a1, b01, acc0);
    acc1 = wmma_bf16(a0, b10, acc1);
    acc1 = wmma_bf16(a1, b11, acc1);
    __syncthreads();
  }

  const int lane = tid & 31;
  const int n_lo = n0 + wn + (lane & 15);
  const int n_hi = n_lo + 16;
  const int mo = (lane & 16) ? 8 : 0;
  const float bv0 = bias[n_lo];
  const float bv1 = bias[n_hi];
#pragma unroll
  for (int r = 0; r < 8; ++r) {
    const size_t m = (size_t)(m0 + wm + mo + r);
    if (BF16OUT) {
      unsigned short* C = (unsigned short*)Cout;
      C[m * N + n_lo] = f2bf(acc0[r] + bv0);
      C[m * N + n_hi] = f2bf(acc1[r] + bv1);
    } else {
      float* C = (float*)Cout;
      C[m * N + n_lo] = acc0[r] + bv0;
      C[m * N + n_hi] = acc1[r] + bv1;
    }
  }
}

// ---------- flash attention over bf16 qkv buffer [4096, 3072] ----------
// grid = (S/64 query blocks, H=16, B=2); hd = 64; scale = 1/sqrt(64) = 0.125.

__global__ __launch_bounds__(256) void attn_flash_wmma(
    const unsigned short* __restrict__ qkv, unsigned short* __restrict__ out) {
  const int S = 2048, D3 = 3072, D = 1024, HD = 64;
  const int LDV = 72;                 // padded ld for transposed V tile
  const int qb = blockIdx.x, h = blockIdx.y, b = blockIdx.z;

  __shared__ unsigned short Qs[64 * 64];
  __shared__ unsigned short Ks[64 * 64];
  __shared__ unsigned short Vt[64 * LDV];     // V transposed: [d][t]
  __shared__ float          Ss[64 * 64];
  __shared__ unsigned short Ps[64 * 64];
  __shared__ float rmax[64], rsum[64], ralpha[64];
  __shared__ float pmx[64 * 4], psm[64 * 4];

  const int tid  = threadIdx.x;
  const int w    = tid >> 5;
  const int lane = tid & 31;
  const int wm = (w >> 1) * 16;
  const int wn = (w & 1) * 32;
  const int row = tid >> 2;           // 0..63
  const int qd  = tid & 3;            // quarter within row
  const int s0  = qd * 8;             // two 16B chunks per 128B row
  const int s1  = s0 + 32;

  const size_t qrow = (size_t)(b * S + qb * 64 + row) * D3 + h * HD;
  // stage Q block (async bf16 copy)
  async_copy16(lds_off_of(&Qs[row * 64 + s0]), qkv + qrow + s0);
  async_copy16(lds_off_of(&Qs[row * 64 + s1]), qkv + qrow + s1);

  if (tid < 64) { rmax[tid] = -3.0e38f; rsum[tid] = 0.0f; }

  const unsigned int k_lds[2] = {lds_off_of(&Ks[row * 64 + s0]),
                                 lds_off_of(&Ks[row * 64 + s1])};

  v8f o0 = {}, o1 = {};

  for (int j = 0; j < 32; ++j) {
    const size_t krow = (size_t)(b * S + j * 64 + row) * D3 + D + h * HD;
    // K: async straight copy
    async_copy16(k_lds[0], qkv + krow + s0);
    async_copy16(k_lds[1], qkv + krow + s1);
    // V: b128 load + transposed b16 stores into Vt[d][t]
    {
      const unsigned short* vg = qkv + krow + D;   // V row t = row
      union { uint4 u4; unsigned short u[8]; } c0, c1;
      c0.u4 = *(const uint4*)(vg + s0);
      c1.u4 = *(const uint4*)(vg + s1);
#pragma unroll
      for (int i = 0; i < 8; ++i) {
        Vt[(s0 + i) * LDV + row] = c0.u[i];
        Vt[(s1 + i) * LDV + row] = c1.u[i];
      }
    }
    wait_async_0();                   // covers Q on j==0, K always
    __syncthreads();

    // S = Q @ K^T  (Ks tile is [n][k]: n = key index, k = d)
    v8f sc0 = {}, sc1 = {};
#pragma unroll
    for (int k0 = 0; k0 < 64; k0 += 32) {
      v16bf a   = load_a_frag(Qs, 64, wm, k0);
      v16bf bk0 = load_b_frag_nk(Ks, 64, wn, k0);
      v16bf bk1 = load_b_frag_nk(Ks, 64, wn + 16, k0);
      sc0 = wmma_bf16(a, bk0, sc0);
      sc1 = wmma_bf16(a, bk1, sc1);
    }
    {
      const int n_lo = wn + (lane & 15), n_hi = n_lo + 16;
      const int mo = (lane & 16) ? 8 : 0;
#pragma unroll
      for (int r = 0; r < 8; ++r) {
        const int m = wm + mo + r;
        Ss[m * 64 + n_lo] = sc0[r] * 0.125f;
        Ss[m * 64 + n_hi] = sc1[r] * 0.125f;
      }
    }
    __syncthreads();

    // ---- online softmax, 4 threads per row ----
    {
      float m4 = -3.0e38f;
#pragma unroll
      for (int c = 0; c < 16; ++c) m4 = fmaxf(m4, Ss[row * 64 + qd * 16 + c]);
      pmx[row * 4 + qd] = m4;
    }
    __syncthreads();
    if (tid < 64) {
      const float mold = rmax[tid];
      float mx = fmaxf(fmaxf(pmx[tid * 4], pmx[tid * 4 + 1]),
                       fmaxf(pmx[tid * 4 + 2], fmaxf(pmx[tid * 4 + 3], mold)));
      rmax[tid]   = mx;
      ralpha[tid] = __expf(mold - mx);
    }
    __syncthreads();
    {
      const float mx = rmax[row];
      float sum4 = 0.0f;
#pragma unroll
      for (int c = 0; c < 16; ++c) {
        const int idx = row * 64 + qd * 16 + c;
        const float p = __expf(Ss[idx] - mx);
        sum4 += p;
        Ps[idx] = f2bf(p);
      }
      psm[row * 4 + qd] = sum4;
    }
    // rescale running O by alpha (valid after the sync above)
    {
      const int mo = (lane & 16) ? 8 : 0;
#pragma unroll
      for (int r = 0; r < 8; ++r) {
        const float al = ralpha[wm + mo + r];
        o0[r] *= al;
        o1[r] *= al;
      }
    }
    __syncthreads();
    if (tid < 64) {
      rsum[tid] = ralpha[tid] * rsum[tid] +
                  (psm[tid * 4] + psm[tid * 4 + 1] + psm[tid * 4 + 2] + psm[tid * 4 + 3]);
    }

    // O += P @ V : A = Ps [q][t], B = Vt [d][t] (both K-contiguous)
#pragma unroll
    for (int k0 = 0; k0 < 64; k0 += 32) {
      v16bf a   = load_a_frag(Ps, 64, wm, k0);
      v16bf bv0 = load_b_frag_nk(Vt, LDV, wn, k0);
      v16bf bv1 = load_b_frag_nk(Vt, LDV, wn + 16, k0);
      o0 = wmma_bf16(a, bv0, o0);
      o1 = wmma_bf16(a, bv1, o1);
    }
    __syncthreads();
  }

  // normalize by 1/l and store bf16 to [B*S, D] with head offset
  {
    const int n_lo = wn + (lane & 15), n_hi = n_lo + 16;
    const int mo = (lane & 16) ? 8 : 0;
#pragma unroll
    for (int r = 0; r < 8; ++r) {
      const int m = wm + mo + r;
      const float inv = 1.0f / rsum[m];
      const size_t grow = (size_t)(b * S + qb * 64 + m);
      out[grow * D + h * HD + n_lo] = f2bf(o0[r] * inv);
      out[grow * D + h * HD + n_hi] = f2bf(o1[r] * inv);
    }
  }
}

// ---------- launch ----------

extern "C" void kernel_launch(void* const* d_in, const int* in_sizes, int n_in,
                              void* d_out, int out_size, void* d_ws, size_t ws_size,
                              hipStream_t stream) {
  const float* q        = (const float*)d_in[0];
  // d_in[1] = k, d_in[2] = v : unused by the reference's self-attention path
  const float* in_w     = (const float*)d_in[3];
  const float* qkv_bias = (const float*)d_in[4];
  const float* out_w    = (const float*)d_in[5];
  const float* out_b    = (const float*)d_in[6];
  float* out = (float*)d_out;

  const int BS = 4096;   // B*S
  const int D  = 1024;
  const int D3 = 3072;

  // workspace layout (bf16 elements)
  unsigned short* qbf     = (unsigned short*)d_ws;              // [4096,1024]
  unsigned short* wqkvbf  = qbf    + (size_t)BS * D;            // [3072,1024]
  unsigned short* woutbf  = wqkvbf + (size_t)D3 * D;            // [1024,1024]
  unsigned short* qkvbf   = woutbf + (size_t)D * D;             // [4096,3072]
  unsigned short* attnbf  = qkvbf  + (size_t)BS * D3;           // [4096,1024]

  // 0) one-time fp32 -> bf16 conversions
  f32_to_bf16<<<(BS * D / 8 + 255) / 256, 256, 0, stream>>>(q, qbf, BS * D / 8);
  f32_to_bf16<<<(D3 * D / 8 + 255) / 256, 256, 0, stream>>>(in_w, wqkvbf, D3 * D / 8);
  f32_to_bf16<<<(D * D / 8 + 255) / 256, 256, 0, stream>>>(out_w, woutbf, D * D / 8);

  // 1) fused QKV projection: qkv = q @ in_w^T + qkv_bias   (bf16 out)
  gemm_bf16_async<true><<<dim3(D3 / 64, BS / 64), 256, 0, stream>>>(
      qbf, wqkvbf, qkv_bias, qkvbf, BS, D3, D);

  // 2) flash attention per (query-block, head, batch)
  attn_flash_wmma<<<dim3(2048 / 64, 16, 2), 256, 0, stream>>>(qkvbf, attnbf);

  // 3) output projection: out = attn @ out_w^T + out_b     (f32 out)
  gemm_bf16_async<false><<<dim3(D / 64, BS / 64), 256, 0, stream>>>(
      attnbf, woutbf, out_b, out, BS, D, D);
}
